// GaussianUpsampling_37357625541235
// MI455X (gfx1250) — matrix operand order repaired and checked by reading, not compiled, not run on hardware
//
#include <hip/hip_runtime.h>
#include <hip/hip_bf16.h>

typedef __attribute__((ext_vector_type(16))) _Float16 v16h;
typedef __attribute__((ext_vector_type(8)))  _Float16 v8h;
typedef __attribute__((ext_vector_type(8)))  float    v8f;
typedef __attribute__((ext_vector_type(4)))  float    v4f;

#define S_IN  512
#define B_SZ  64
#define D_SZ  512
#define T_OUT 2048
#define SP    520   // padded LDS row stride in halves (bank-conflict-free A loads)

// ---------------- kernel 1: per-batch cumsum -> centers, 1/sigma, log(sigma), seq_len
__global__ void gu_setup(const int* __restrict__ dur, const float* __restrict__ rng,
                         float* __restrict__ c, float* __restrict__ invs,
                         float* __restrict__ logs, float* __restrict__ seq) {
    int b = threadIdx.x;
    if (b >= B_SZ) return;
    int cum = 0;
    for (int s = 0; s < S_IN; ++s) {
        int d = dur[s * B_SZ + b];          // coalesced across the 64 lanes
        cum += d;
        float cc = (float)cum - 0.5f * (float)d;
        float sg = rng[s * B_SZ + b];
        c[b * S_IN + s]    = cc;
        invs[b * S_IN + s] = 1.0f / sg;
        logs[b * S_IN + s] = __logf(sg);
    }
    seq[b] = (float)cum;                    // sequence_lengths output
}

// ---------------- kernel 2: x (S,B,D) f32 -> x16 [b][d][s] f16 (tiled transpose)
__global__ __launch_bounds__(256)
void gu_xpose(const float* __restrict__ x, _Float16* __restrict__ x16) {
    __shared__ _Float16 tile[32][33];
    const int b  = blockIdx.z;
    const int d0 = blockIdx.x * 32;
    const int s0 = blockIdx.y * 32;
    const int lane = threadIdx.x & 31, w = threadIdx.x >> 5;
#pragma unroll
    for (int i = 0; i < 4; ++i) {
        int sl = w + 8 * i;
        float v = x[((size_t)(s0 + sl) * B_SZ + b) * D_SZ + d0 + lane]; // coalesced
        tile[sl][lane] = (_Float16)v;
    }
    __syncthreads();
#pragma unroll
    for (int i = 0; i < 4; ++i) {
        int dl = w + 8 * i;
        // 32 consecutive s per wave -> 64B contiguous stores
        x16[((size_t)b * D_SZ + d0 + dl) * S_IN + s0 + lane] = tile[lane][dl];
    }
}

// ---------------- kernel 3: fused softmax + WMMA GEMM per (b, 32-t tile)
__global__ __launch_bounds__(256)
void gu_main(const float* __restrict__ c_all, const float* __restrict__ invs_all,
             const float* __restrict__ logs_all, const _Float16* __restrict__ x16,
             float* __restrict__ out1, float* __restrict__ wout) {
    __shared__ _Float16 w16s[32 * SP];
    __shared__ float maxv[32];
    __shared__ float rsum[32];

    const int b   = blockIdx.y;
    const int t0  = blockIdx.x * 32;
    const int tid = threadIdx.x;
    const float* cb = c_all   + b * S_IN;
    const float* ib = invs_all + b * S_IN;
    const float* lb = logs_all + b * S_IN;

    // ---- phase 1: softmax stats (8 lanes cooperate per t; -0.5*log(2pi) cancels)
    {
        const int tl = tid >> 3, g = tid & 7;
        const float t = (float)(t0 + tl);
        float mx = -3.0e38f;
        for (int i = 0; i < 64; ++i) {
            int s = g + 8 * i;
            float z  = (t - cb[s]) * ib[s];
            float lp = -0.5f * z * z - lb[s];
            mx = fmaxf(mx, lp);
        }
        mx = fmaxf(mx, __shfl_xor(mx, 1, 8));
        mx = fmaxf(mx, __shfl_xor(mx, 2, 8));
        mx = fmaxf(mx, __shfl_xor(mx, 4, 8));
        float sm = 0.0f;
        for (int i = 0; i < 64; ++i) {
            int s = g + 8 * i;
            float z  = (t - cb[s]) * ib[s];
            float lp = -0.5f * z * z - lb[s];
            sm += __expf(lp - mx);
        }
        sm += __shfl_xor(sm, 1, 8);
        sm += __shfl_xor(sm, 2, 8);
        sm += __shfl_xor(sm, 4, 8);
        if (g == 0) { maxv[tl] = mx; rsum[tl] = 1.0f / sm; }
    }
    __syncthreads();

    // ---- phase 2: weights -> f32 global (S,B,T) + f16 LDS [t][s]
    {
        const int tq = (tid & 7) * 4;
        for (int j = 0; j < 16; ++j) {
            int s = (tid >> 3) + 32 * j;
            float cc = cb[s], is = ib[s], ls = lb[s];
            v4f wv;
#pragma unroll
            for (int k = 0; k < 4; ++k) {
                float t  = (float)(t0 + tq + k);
                float z  = (t - cc) * is;
                float lp = -0.5f * z * z - ls;
                float wv1 = __expf(lp - maxv[tq + k]) * rsum[tq + k];
                wv[k] = wv1;
                w16s[(tq + k) * SP + s] = (_Float16)wv1;
            }
            // 8-lane groups write 128B contiguous per s-row
            *(v4f*)&wout[((size_t)s * B_SZ + b) * T_OUT + t0 + tq] = wv;
        }
    }
    __syncthreads();

    // ---- phase 3: GEMM  out[t0..t0+31][b][:] = W(32xS) * X(SxD)  via f16 WMMA
    {
        const int w  = tid >> 5;
        const int L  = tid & 31;
        const int Ln = L & 15;          // N / M within tile
        const int Lh = L >> 4;          // lane-half select
        const _Float16* xb = x16 + (size_t)b * D_SZ * S_IN;

        for (int j = 0; j < 8; ++j) {               // 8 tile-jobs per wave
            int jj = w * 8 + j;
            int nt = jj >> 1, mt = jj & 1;
            int d0 = nt * 16;
            v8f acc = {};
            // A (16x32 f16): lanes 0-15 K in {0..7,16..23}; lanes 16-31 K in {8..15,24..31}
            const _Float16* ap = &w16s[(mt * 16 + Ln) * SP + Lh * 8];
            // B (32x16 f16): lane = column d; lanes 0-15 K in {0..7,16..23}; lanes 16-31 K in {8..15,24..31}
            const _Float16* bp = xb + (size_t)(d0 + Ln) * S_IN + Lh * 8;
#pragma unroll 4
            for (int k0 = 0; k0 < S_IN; k0 += 32) {
                v8h alo = *(const v8h*)(ap + k0);
                v8h ahi = *(const v8h*)(ap + k0 + 16);
                v16h A = __builtin_shufflevector(alo, ahi,
                         0,1,2,3,4,5,6,7,8,9,10,11,12,13,14,15);
                v8h blo = *(const v8h*)(bp + k0);
                v8h bhi = *(const v8h*)(bp + k0 + 16);
                v16h Bm = __builtin_shufflevector(blo, bhi,
                         0,1,2,3,4,5,6,7,8,9,10,11,12,13,14,15);
                acc = __builtin_amdgcn_wmma_f32_16x16x32_f16(
                          false, A, false, Bm, (short)0, acc, false, false);
            }
            // C/D layout: VGPR r -> row M = r (lanes 0-15) or r+8 (lanes 16-31)
#pragma unroll
            for (int r = 0; r < 8; ++r) {
                int t = t0 + mt * 16 + Lh * 8 + r;
                out1[((size_t)t * B_SZ + b) * D_SZ + d0 + Ln] = acc[r];
            }
        }
    }
}

extern "C" void kernel_launch(void* const* d_in, const int* in_sizes, int n_in,
                              void* d_out, int out_size, void* d_ws, size_t ws_size,
                              hipStream_t stream) {
    (void)in_sizes; (void)n_in; (void)out_size; (void)ws_size;
    const float* x   = (const float*)d_in[0];   // (S,B,D) f32
    const int*   dur = (const int*)d_in[1];     // (S,B) i32
    const float* rng = (const float*)d_in[2];   // (S,B) f32
    // d_in[3] = sequence_out (static 2048, hardcoded)

    float* out  = (float*)d_out;
    float* out1 = out;                                          // (T,B,D)
    float* seq  = out + (size_t)T_OUT * B_SZ * D_SZ;            // (B,)
    float* wout = seq + B_SZ;                                   // (S,B,T)

    float* wsf   = (float*)d_ws;
    float* c     = wsf;
    float* invs  = wsf + (size_t)B_SZ * S_IN;
    float* logs  = wsf + (size_t)2 * B_SZ * S_IN;
    _Float16* x16 = (_Float16*)(wsf + (size_t)3 * B_SZ * S_IN); // 16B-aligned, ~34MB

    gu_setup<<<1, 64, 0, stream>>>(dur, rng, c, invs, logs, seq);
    gu_xpose<<<dim3(D_SZ / 32, S_IN / 32, B_SZ), 256, 0, stream>>>(x, x16);
    gu_main<<<dim3(T_OUT / 32, B_SZ), 256, 0, stream>>>(c, invs, logs, x16, out1, wout);
}
